// FP8BatchMatMulDense_34720515621130
// MI455X (gfx1250) — 8-bit matrix operand order repaired and checked by reading, compiled, not run on hardware
//
#include <hip/hip_runtime.h>
#include <cmath>

typedef __attribute__((ext_vector_type(16))) int   v16i;
typedef __attribute__((ext_vector_type(8)))  float v8f;

#define FP8_MAX 448.0f

#if defined(__AMDGCN__) && __has_builtin(__builtin_amdgcn_cvt_pk_fp8_f32)
#define HAVE_HW_FP8 1
#else
#define HAVE_HW_FP8 0
#endif


__device__ __forceinline__ unsigned char f32_to_e4m3_sw(float v) {
  unsigned int u = __float_as_uint(v);
  unsigned int s = (u >> 24) & 0x80u;
  float ax = fabsf(v);
  if (ax >= 448.0f) return (unsigned char)(s | 0x7E);
  if (ax < 0.015625f) {
    int m = (int)rintf(ax * 512.0f);
    if (m > 7) return (unsigned char)(s | 0x08);
    return (unsigned char)(s | (unsigned)m);
  }
  unsigned int au = __float_as_uint(ax);
  unsigned int lsb = (au >> 20) & 1u;
  au += 0x7FFFFu + lsb;
  int e = (int)((au >> 23) & 0xFFu) - 127;
  unsigned int m = (au >> 20) & 0x7u;
  int ef = e + 7;
  if (ef > 15 || (ef == 15 && m > 6)) return (unsigned char)(s | 0x7E);
  return (unsigned char)(s | ((unsigned)ef << 3) | m);
}

__device__ __forceinline__ unsigned char fq1(float x, float scale) {
  float v = fminf(fmaxf(x * scale, -FP8_MAX), FP8_MAX);
#if HAVE_HW_FP8
  return (unsigned char)(__builtin_amdgcn_cvt_pk_fp8_f32(v, v, 0, false) & 0xff);
#else
  return f32_to_e4m3_sw(v);
#endif
}


__global__ __launch_bounds__(256) void amax_partial_kernel(
    const float* __restrict__ x, unsigned long long n, float* __restrict__ partial) {
  __shared__ float sm[256];
  float m = 0.0f;
  for (unsigned long long i = (unsigned long long)blockIdx.x * 256ull + threadIdx.x;
       i < n; i += (unsigned long long)gridDim.x * 256ull)
    m = fmaxf(m, fabsf(x[i]));
  sm[threadIdx.x] = m;
  __syncthreads();
  for (int s = 128; s > 0; s >>= 1) {
    if ((int)threadIdx.x < s) sm[threadIdx.x] = fmaxf(sm[threadIdx.x], sm[threadIdx.x + s]);
    __syncthreads();
  }
  if (threadIdx.x == 0) partial[blockIdx.x] = sm[0];
}

__global__ __launch_bounds__(256) void finalize_scales_kernel(
    const float* __restrict__ px, const float* __restrict__ pw, float* __restrict__ scales) {
  __shared__ float sx[256];
  __shared__ float sw_[256];
  const int t = threadIdx.x;
  sx[t]  = fmaxf(px[t], px[t + 256]);
  sw_[t] = fmaxf(pw[t], pw[t + 256]);
  __syncthreads();
  for (int s = 128; s > 0; s >>= 1) {
    if (t < s) {
      sx[t]  = fmaxf(sx[t],  sx[t + s]);
      sw_[t] = fmaxf(sw_[t], sw_[t + s]);
    }
    __syncthreads();
  }
  if (t == 0) {
    float ax = fmaxf(sx[0],  1e-12f);
    float aw = fmaxf(sw_[0], 1e-12f);
    float scx = FP8_MAX / ax;
    float scw = FP8_MAX / aw;
    scales[0] = scx;
    scales[1] = scw;
    scales[2] = 1.0f / (scx * scw);
  }
}


__global__ __launch_bounds__(256) void quant_x_kernel(
    const float* __restrict__ x, const float* __restrict__ scales,
    unsigned int* __restrict__ xq) {
  const float s = scales[0];
  const size_t i = (size_t)blockIdx.x * 256 + threadIdx.x;
  const float4 v = ((const float4*)x)[i];
  unsigned int r = (unsigned int)fq1(v.x, s)
                 | ((unsigned int)fq1(v.y, s) << 8)
                 | ((unsigned int)fq1(v.z, s) << 16)
                 | ((unsigned int)fq1(v.w, s) << 24);
  xq[i] = r;
}


__global__ __launch_bounds__(256) void quant_w_kernel(
    const float* __restrict__ w, const float* __restrict__ scales,
    unsigned char* __restrict__ wt) {
  __shared__ __align__(16) unsigned char tile[64 * 64];
  const float s = scales[1];
  const int t  = threadIdx.x;
  const int tn = blockIdx.x & 31;
  const int tk = (blockIdx.x >> 5) & 31;
  const int b  = blockIdx.x >> 10;
  const size_t base = (size_t)b * 2048 * 2048 + (size_t)tk * 64 * 2048 + (size_t)tn * 64;
#pragma unroll
  for (int i = 0; i < 16; ++i) {
    const int flat = t + i * 256;
    const int kl = flat >> 6;
    const int nl = flat & 63;
    const float v = w[base + (size_t)kl * 2048 + nl];
    tile[nl * 64 + kl] = fq1(v, s);
  }
  __syncthreads();
  const int nl2 = t >> 2;
  const int kc  = (t & 3) * 16;
  const uint4 q = *(const uint4*)&tile[nl2 * 64 + kc];
  *(uint4*)(wt + (size_t)b * 2048 * 2048
               + (size_t)(tn * 64 + nl2) * 2048
               + (size_t)(tk * 64 + kc)) = q;
}


__global__ __launch_bounds__(128) void fp8_bmm_wmma_kernel(
    const unsigned char* __restrict__ Xq, const unsigned char* __restrict__ Wt,
    const float* __restrict__ bias, const float* __restrict__ scales,
    float* __restrict__ out) {
  const int lane = threadIdx.x & 31;
  const int wid  = blockIdx.x * 4 + (threadIdx.x >> 5);
  const int tn = wid & 31;
  const int tm = (wid >> 5) & 7;
  const int b  = wid >> 8;
  const int lr = lane & 15;
  const int lh = lane >> 4;

  const unsigned char* Ab = Xq + ((size_t)b * 512 + tm * 64 + lr) * 2048 + (size_t)lh * 16;
  const unsigned char* Bb = Wt + ((size_t)b * 2048 + tn * 64 + lr) * 2048 + (size_t)lh * 16;

  v8f acc[4][4] = {};

  for (int k0 = 0; k0 < 2048; k0 += 128) {
    v16i A[4], Bv[4];
#pragma unroll
    for (int mi = 0; mi < 4; ++mi) {
      const unsigned char* p = Ab + (size_t)mi * 16 * 2048 + k0;
#pragma unroll
      for (int j = 0; j < 8; ++j) {
        const uint2 d = *(const uint2*)(p + 32 * (j >> 1) + 8 * (j & 1));
        A[mi][2 * j]     = (int)d.x;
        A[mi][2 * j + 1] = (int)d.y;
      }
    }
#pragma unroll
    for (int ni = 0; ni < 4; ++ni) {
      const unsigned char* p = Bb + (size_t)ni * 16 * 2048 + k0;
#pragma unroll
      for (int g = 0; g < 4; ++g) {
        const uint4 q = *(const uint4*)(p + 32 * g);
        Bv[ni][4 * g]     = (int)q.x;
        Bv[ni][4 * g + 1] = (int)q.y;
        Bv[ni][4 * g + 2] = (int)q.z;
        Bv[ni][4 * g + 3] = (int)q.w;
      }
    }
#pragma unroll
    for (int mi = 0; mi < 4; ++mi)
#pragma unroll
      for (int ni = 0; ni < 4; ++ni)
        acc[mi][ni] = __builtin_amdgcn_wmma_f32_16x16x128_fp8_fp8(
            A[mi], Bv[ni], (short)0, acc[mi][ni], false, false);
  }

  const float inv = scales[2];
#pragma unroll
  for (int ni = 0; ni < 4; ++ni) {
    const int n = tn * 64 + ni * 16 + lr;
    const float bv = bias[(size_t)b * 2048 + n];
#pragma unroll
    for (int mi = 0; mi < 4; ++mi) {
      const int m0 = tm * 64 + mi * 16 + lh * 8;
      float* o = out + ((size_t)b * 512 + m0) * 2048 + n;
#pragma unroll
      for (int r = 0; r < 8; ++r)
        o[(size_t)r * 2048] = acc[mi][ni][r] * inv + bv;
    }
  }
}


extern "C" void kernel_launch(void* const* d_in, const int* in_sizes, int n_in,
                              void* d_out, int out_size, void* d_ws, size_t ws_size,
                              hipStream_t stream) {
  const float* x    = (const float*)d_in[0];
  const float* w    = (const float*)d_in[1];
  const float* bias = (const float*)d_in[2];
  float* out = (float*)d_out;
  char* ws = (char*)d_ws;

  float* scales = (float*)ws;
  float* px = (float*)(ws + 256);
  float* pw = (float*)(ws + 256 + 2048);
  unsigned char* xq = (unsigned char*)(ws + 8192);
  unsigned char* wt = xq + (size_t)32 * 512 * 2048;

  const unsigned long long nx = 32ull * 512 * 2048;
  const unsigned long long nw = 32ull * 2048 * 2048;

  amax_partial_kernel<<<512, 256, 0, stream>>>(x, nx, px);
  amax_partial_kernel<<<512, 256, 0, stream>>>(w, nw, pw);
  finalize_scales_kernel<<<1, 256, 0, stream>>>(px, pw, scales);
  quant_x_kernel<<<(unsigned)(nx / 4 / 256), 256, 0, stream>>>(x, scales, (unsigned int*)xq);
  quant_w_kernel<<<32 * 32 * 32, 256, 0, stream>>>(w, scales, wt);
  fp8_bmm_wmma_kernel<<<2048, 128, 0, stream>>>(xq, wt, bias, scales, out);
}
